// DifferentialMultiHeadAttention_52020643889367
// MI455X (gfx1250) — matrix operand order repaired and checked by reading, not compiled, not run on hardware
//
#include <hip/hip_runtime.h>
#include <hip/hip_bf16.h>
#include <math.h>

// ---------------------------------------------------------------------------
// Differential Multi-Head Attention for MI455X (gfx1250, wave32, WMMA)
//   B=2, S=2048, HIDDEN=1024, HEADS=16, HEAD_DIM=64, DIFF_DIM=32
// All GEMMs run on v_wmma_f32_16x16x32_f16 (f16 in, f32 accumulate).
// Projection GEMMs are register-blocked 4x2 (64x32 macro-tile per wave) so
// each K-step issues 8 WMMAs on 12 b128 loads. Attention is flash-style with
// dual online-softmax accumulators combined as a1 - lambda*a2.
// ---------------------------------------------------------------------------

#define S_LEN     2048
#define HIDDEN_D  1024
#define HEADS_N   16
#define HEAD_DIM  64
#define DIFF_DIM  32
#define LAMBDA_INIT_F 0.2f
#define INV_SCALE 0.17677669529663687f   // 1/sqrt(32)
#define LN_EPS_F  1e-5f
#define M_ROWS    4096                    // B*S

typedef __attribute__((ext_vector_type(16))) _Float16 v16h;
typedef __attribute__((ext_vector_type(8)))  _Float16 v8h;
typedef __attribute__((ext_vector_type(8)))  float    v8f;

// --- WMMA fragment loaders ---------------------------------------------------
// A fragment (16x32 f16, M x K). Lane L holds row M=L%16.
//   lanes 0-15 : elems 0..7 = K 0..7,   elems 8..15 = K 16..23
//   lanes 16-31: elems 0..7 = K 8..15,  elems 8..15 = K 24..31
__device__ __forceinline__ v16h load_a_frag(const _Float16* __restrict__ row,
                                            int kbase, int lane) {
  int hi = (lane & 16) ? 8 : 0;
  v8h lo = *(const v8h*)(row + kbase + hi);
  v8h hp = *(const v8h*)(row + kbase + hi + 16);
  v16h r;
#pragma unroll
  for (int i = 0; i < 8; ++i) { r[i] = lo[i]; r[8 + i] = hp[i]; }
  return r;
}

// B fragment (32x16 f16, K x N). Lane L holds column N=L%16.
//   lanes 0-15 : elems 0..7 = K 0..7,   elems 8..15 = K 16..23
//   lanes 16-31: elems 0..7 = K 8..15,  elems 8..15 = K 24..31
// `col` points at element K=0 of the lane's column (column stored K-contiguous).
__device__ __forceinline__ v16h load_b_frag(const _Float16* __restrict__ col,
                                            int lane) {
  int hi = (lane & 16) ? 8 : 0;
  v8h a = *(const v8h*)(col + hi);
  v8h b = *(const v8h*)(col + hi + 16);
  v16h r;
#pragma unroll
  for (int i = 0; i < 8; ++i) { r[i] = a[i]; r[8 + i] = b[i]; }
  return r;
}

__device__ __forceinline__ v8f wmma_f16(v16h a, v16h b, v8f c) {
  return __builtin_amdgcn_wmma_f32_16x16x32_f16(false, a, false, b,
                                                (short)0, c, false, false);
}

// --- conversion kernels ------------------------------------------------------
__global__ void cvt_f32_to_f16(const float* __restrict__ in,
                               _Float16* __restrict__ out, int n) {
  int i = blockIdx.x * 256 + threadIdx.x;
  if (i < n) out[i] = (_Float16)in[i];
}

// WT[n*1024 + k] = (f16) W[k*1024 + n]
__global__ void transpose_w(const float* __restrict__ W,
                            _Float16* __restrict__ WT) {
  int i = blockIdx.x * 256 + threadIdx.x;  // 0 .. 1024*1024-1
  int n = i >> 10, k = i & 1023;
  WT[(size_t)n * 1024 + k] = (_Float16)W[(size_t)k * 1024 + n];
}

// --- QKV projection: q/k/v = x @ W* + b* ------------------------------------
// One wave computes a 64x32 macro-tile (4 M-tiles x 2 N-tiles, register
// blocked). which: 0=q,1=k,2=v.
// q,k stored as [b][half][h][s][32]; v stored transposed [b][h][d][s].
__global__ __launch_bounds__(128) void qkv_gemm(
    const _Float16* __restrict__ xh,
    const _Float16* __restrict__ WqT, const _Float16* __restrict__ WkT,
    const _Float16* __restrict__ WvT,
    const float* __restrict__ bq, const float* __restrict__ bk,
    const float* __restrict__ bv,
    _Float16* __restrict__ qh, _Float16* __restrict__ kh,
    _Float16* __restrict__ vTh) {
  int lane = threadIdx.x & 31;
  int wave = threadIdx.x >> 5;
  int tile = blockIdx.x * 4 + wave;          // 3 * 64 * 32 macro-tiles
  int which = tile / (64 * 32);
  int rem   = tile % (64 * 32);
  int mblk  = rem >> 5;                      // 64 macro M blocks (64 rows)
  int nblk  = rem & 31;                      // 32 macro N blocks (32 cols)

  const _Float16* WT  = (which == 0) ? WqT : (which == 1) ? WkT : WvT;
  const float*    bia = (which == 0) ? bq  : (which == 1) ? bk  : bv;

  int c16 = lane & 15, g = lane >> 4;
  int m0 = mblk * 64;
  int n0 = nblk * 32;

  const _Float16* arow[4];
#pragma unroll
  for (int i = 0; i < 4; ++i)
    arow[i] = xh + (size_t)(m0 + i * 16 + c16) * HIDDEN_D;
  const _Float16* bcol0 = WT + (size_t)(n0 + c16) * HIDDEN_D;
  const _Float16* bcol1 = WT + (size_t)(n0 + 16 + c16) * HIDDEN_D;

  v8f acc[4][2] = {};
#pragma unroll 2
  for (int kb = 0; kb < HIDDEN_D; kb += 32) {
    v16h bf0 = load_b_frag(bcol0 + kb, lane);
    v16h bf1 = load_b_frag(bcol1 + kb, lane);
#pragma unroll
    for (int i = 0; i < 4; ++i) {
      v16h af = load_a_frag(arow[i], kb, lane);
      acc[i][0] = wmma_f16(af, bf0, acc[i][0]);
      acc[i][1] = wmma_f16(af, bf1, acc[i][1]);
    }
  }

#pragma unroll
  for (int jn = 0; jn < 2; ++jn) {
    int ncol = n0 + jn * 16 + c16;
    float bval = bia[ncol];
#pragma unroll
    for (int i = 0; i < 4; ++i) {
#pragma unroll
      for (int r = 0; r < 8; ++r) {
        int m = m0 + i * 16 + r + 8 * g;
        int bb = m >> 11;                 // batch
        int s  = m & (S_LEN - 1);         // sequence index
        _Float16 hv = (_Float16)(acc[i][jn][r] + bval);
        if (which == 2) {                 // V, transposed per head: [b][h][d][s]
          int hh = ncol >> 6, dd = ncol & 63;
          vTh[(((size_t)bb * HEADS_N + hh) * HEAD_DIM + dd) * S_LEN + s] = hv;
        } else {                          // Q/K split halves: [b][half][h][s][32]
          int half = ncol >> 9, hh = (ncol >> 5) & 15, dd = ncol & 31;
          _Float16* dst = (which == 0) ? qh : kh;
          dst[((((size_t)bb * 2 + half) * HEADS_N + hh) * S_LEN + s) * DIFF_DIM + dd] = hv;
        }
      }
    }
  }
}

// --- flash differential attention -------------------------------------------
// One wave per (b, h, 16-row query tile). Streams keys in 32-wide tiles.
__global__ __launch_bounds__(128) void diff_attn(
    const _Float16* __restrict__ qh, const _Float16* __restrict__ kh,
    const _Float16* __restrict__ vTh, const int* __restrict__ mask,
    const float* __restrict__ lq1, const float* __restrict__ lk1,
    const float* __restrict__ lq2, const float* __restrict__ lk2,
    const float* __restrict__ ln_g, const float* __restrict__ ln_b,
    _Float16* __restrict__ aoh) {
  __shared__ __align__(16) _Float16 lds[4 * 1024];  // per wave: p1,p2 16x32 tiles

  int lane = threadIdx.x & 31;
  int wave = threadIdx.x >> 5;
  int tile = blockIdx.x * 4 + wave;      // 2*16*128 = 4096 tiles exactly
  int mtile = tile & 127;
  int h = (tile >> 7) & 15;
  int b = tile >> 11;
  int c16 = lane & 15, g = lane >> 4;

  // lambda = exp(sum lq1*lk1) - exp(sum lq2*lk2) + lambda_init  (wave32 butterfly)
  float t1 = lq1[lane] * lk1[lane];
  float t2 = lq2[lane] * lk2[lane];
#pragma unroll
  for (int o = 16; o >= 1; o >>= 1) {
    t1 += __shfl_xor(t1, o, 32);
    t2 += __shfl_xor(t2, o, 32);
  }
  float lam = __expf(t1) - __expf(t2) + LAMBDA_INIT_F;

  // Q fragments (held for the whole key loop)
  int srow0 = mtile * 16 + c16;
  const _Float16* q1row =
      qh + ((((size_t)b * 2 + 0) * HEADS_N + h) * S_LEN + srow0) * DIFF_DIM;
  const _Float16* q2row =
      qh + ((((size_t)b * 2 + 1) * HEADS_N + h) * S_LEN + srow0) * DIFF_DIM;
  v16h q1f = load_a_frag(q1row, 0, lane);
  v16h q2f = load_a_frag(q2row, 0, lane);

  const size_t kbase1 = (((size_t)b * 2 + 0) * HEADS_N + h) * S_LEN * DIFF_DIM;
  const size_t kbase2 = (((size_t)b * 2 + 1) * HEADS_N + h) * S_LEN * DIFF_DIM;
  const size_t vbase  = ((size_t)b * HEADS_N + h) * HEAD_DIM * (size_t)S_LEN;

  _Float16* p1 = &lds[wave * 1024];
  _Float16* p2 = p1 + 512;

  v8f acc1[4] = {}, acc2[4] = {};
  float m1r[8], l1r[8], m2r[8], l2r[8];
#pragma unroll
  for (int r = 0; r < 8; ++r) {
    m1r[r] = -1e30f; m2r[r] = -1e30f; l1r[r] = 0.f; l2r[r] = 0.f;
  }

  for (int j = 0; j < S_LEN / 32; ++j) {
    int kc0 = j * 32 + c16;
    int kc1 = kc0 + 16;
    v16h k1b0 = load_b_frag(kh + kbase1 + (size_t)kc0 * DIFF_DIM, lane);
    v16h k1b1 = load_b_frag(kh + kbase1 + (size_t)kc1 * DIFF_DIM, lane);
    v16h k2b0 = load_b_frag(kh + kbase2 + (size_t)kc0 * DIFF_DIM, lane);
    v16h k2b1 = load_b_frag(kh + kbase2 + (size_t)kc1 * DIFF_DIM, lane);
    if (j + 1 < S_LEN / 32) {  // prefetch next key tile (global_prefetch)
      __builtin_prefetch(kh + kbase1 + (size_t)(kc0 + 32) * DIFF_DIM, 0, 1);
      __builtin_prefetch(kh + kbase2 + (size_t)(kc0 + 32) * DIFF_DIM, 0, 1);
    }

    v8f z = {};
    v8f s1lo = wmma_f16(q1f, k1b0, z);
    v8f s1hi = wmma_f16(q1f, k1b1, z);
    v8f s2lo = wmma_f16(q2f, k2b0, z);
    v8f s2hi = wmma_f16(q2f, k2b1, z);

#pragma unroll
    for (int r = 0; r < 8; ++r) {
      int row = mtile * 16 + r + 8 * g;
      const int* mrow = mask + ((size_t)b * S_LEN + row) * S_LEN + j * 32;
      float a1 = s1lo[r] * INV_SCALE, b1 = s1hi[r] * INV_SCALE;
      float a2 = s2lo[r] * INV_SCALE, b2 = s2hi[r] * INV_SCALE;
      if (mrow[c16]      == 0) { a1 = -1e9f; a2 = -1e9f; }
      if (mrow[16 + c16] == 0) { b1 = -1e9f; b2 = -1e9f; }

      // online softmax, half 1
      float mx = fmaxf(a1, b1);
#pragma unroll
      for (int o = 8; o >= 1; o >>= 1) mx = fmaxf(mx, __shfl_xor(mx, o, 32));
      float mn = fmaxf(m1r[r], mx);
      float sc = __expf(m1r[r] - mn);
      float pa = __expf(a1 - mn), pb = __expf(b1 - mn);
      float rs = pa + pb;
#pragma unroll
      for (int o = 8; o >= 1; o >>= 1) rs += __shfl_xor(rs, o, 32);
      l1r[r] = l1r[r] * sc + rs;
      m1r[r] = mn;
#pragma unroll
      for (int nt = 0; nt < 4; ++nt) acc1[nt][r] *= sc;
      p1[(r + 8 * g) * 32 + c16]      = (_Float16)pa;
      p1[(r + 8 * g) * 32 + 16 + c16] = (_Float16)pb;

      // online softmax, half 2
      mx = fmaxf(a2, b2);
#pragma unroll
      for (int o = 8; o >= 1; o >>= 1) mx = fmaxf(mx, __shfl_xor(mx, o, 32));
      mn = fmaxf(m2r[r], mx);
      sc = __expf(m2r[r] - mn);
      pa = __expf(a2 - mn); pb = __expf(b2 - mn);
      rs = pa + pb;
#pragma unroll
      for (int o = 8; o >= 1; o >>= 1) rs += __shfl_xor(rs, o, 32);
      l2r[r] = l2r[r] * sc + rs;
      m2r[r] = mn;
#pragma unroll
      for (int nt = 0; nt < 4; ++nt) acc2[nt][r] *= sc;
      p2[(r + 8 * g) * 32 + c16]      = (_Float16)pa;
      p2[(r + 8 * g) * 32 + 16 + c16] = (_Float16)pb;
    }
    __syncthreads();

    // P tiles (C-layout in LDS) -> A fragments, then PV WMMAs
    v16h p1f = load_a_frag(p1 + c16 * 32, 0, lane);
    v16h p2f = load_a_frag(p2 + c16 * 32, 0, lane);
#pragma unroll
    for (int nt = 0; nt < 4; ++nt) {
      int d = nt * 16 + c16;
      const _Float16* vcol = vTh + vbase + (size_t)d * S_LEN + j * 32;
      if (j + 1 < S_LEN / 32) __builtin_prefetch(vcol + 32, 0, 1);
      v16h vf = load_b_frag(vcol, lane);
      acc1[nt] = wmma_f16(p1f, vf, acc1[nt]);
      acc2[nt] = wmma_f16(p2f, vf, acc2[nt]);
    }
    __syncthreads();
  }

  // epilogue: normalize, diff-combine, per-head LayerNorm, scale, store f16
#pragma unroll
  for (int nt = 0; nt < 4; ++nt)
#pragma unroll
    for (int r = 0; r < 8; ++r)
      acc1[nt][r] = acc1[nt][r] / l1r[r] - lam * (acc2[nt][r] / l2r[r]);

#pragma unroll
  for (int r = 0; r < 8; ++r) {
    float s = acc1[0][r] + acc1[1][r] + acc1[2][r] + acc1[3][r];
#pragma unroll
    for (int o = 8; o >= 1; o >>= 1) s += __shfl_xor(s, o, 32);
    float mu = s * (1.0f / 64.0f);
    float vs = 0.f;
#pragma unroll
    for (int nt = 0; nt < 4; ++nt) {
      float d = acc1[nt][r] - mu;
      vs += d * d;
    }
#pragma unroll
    for (int o = 8; o >= 1; o >>= 1) vs += __shfl_xor(vs, o, 32);
    float rstd = rsqrtf(vs * (1.0f / 64.0f) + LN_EPS_F);
    int row = mtile * 16 + r + 8 * g;
#pragma unroll
    for (int nt = 0; nt < 4; ++nt) {
      int d = nt * 16 + c16;
      float nv = (acc1[nt][r] - mu) * rstd;
      float ov = (nv * ln_g[h * 64 + d] + ln_b[h * 64 + d]) * (1.0f - LAMBDA_INIT_F);
      aoh[((size_t)b * S_LEN + row) * HIDDEN_D + h * 64 + d] = (_Float16)ov;
    }
  }
}

// --- output projection: out = aoh @ Wo + bo (fp32 out), 4x2 blocked ---------
__global__ __launch_bounds__(128) void out_proj(
    const _Float16* __restrict__ aoh, const _Float16* __restrict__ WoT,
    const float* __restrict__ bo, float* __restrict__ out) {
  int lane = threadIdx.x & 31;
  int wave = threadIdx.x >> 5;
  int tile = blockIdx.x * 4 + wave;       // 64 * 32 macro-tiles
  int mblk = tile >> 5;
  int nblk = tile & 31;
  int c16 = lane & 15, g = lane >> 4;
  int m0 = mblk * 64;
  int n0 = nblk * 32;

  const _Float16* arow[4];
#pragma unroll
  for (int i = 0; i < 4; ++i)
    arow[i] = aoh + (size_t)(m0 + i * 16 + c16) * HIDDEN_D;
  const _Float16* bcol0 = WoT + (size_t)(n0 + c16) * HIDDEN_D;
  const _Float16* bcol1 = WoT + (size_t)(n0 + 16 + c16) * HIDDEN_D;

  v8f acc[4][2] = {};
#pragma unroll 2
  for (int kb = 0; kb < HIDDEN_D; kb += 32) {
    v16h bf0 = load_b_frag(bcol0 + kb, lane);
    v16h bf1 = load_b_frag(bcol1 + kb, lane);
#pragma unroll
    for (int i = 0; i < 4; ++i) {
      v16h af = load_a_frag(arow[i], kb, lane);
      acc[i][0] = wmma_f16(af, bf0, acc[i][0]);
      acc[i][1] = wmma_f16(af, bf1, acc[i][1]);
    }
  }

#pragma unroll
  for (int jn = 0; jn < 2; ++jn) {
    int ncol = n0 + jn * 16 + c16;
    float bval = bo[ncol];
#pragma unroll
    for (int i = 0; i < 4; ++i) {
#pragma unroll
      for (int r = 0; r < 8; ++r) {
        int m = m0 + i * 16 + r + 8 * g;
        out[(size_t)m * HIDDEN_D + ncol] = acc[i][jn][r] + bval;
      }
    }
  }
}

// ---------------------------------------------------------------------------
extern "C" void kernel_launch(void* const* d_in, const int* in_sizes, int n_in,
                              void* d_out, int out_size, void* d_ws, size_t ws_size,
                              hipStream_t stream) {
  const float* x    = (const float*)d_in[0];
  const int*   mask = (const int*)d_in[1];
  const float* Wq = (const float*)d_in[2];  const float* bq = (const float*)d_in[3];
  const float* Wk = (const float*)d_in[4];  const float* bk = (const float*)d_in[5];
  const float* Wv = (const float*)d_in[6];  const float* bv = (const float*)d_in[7];
  const float* Wo = (const float*)d_in[8];  const float* bo = (const float*)d_in[9];
  const float* lq1 = (const float*)d_in[10]; const float* lk1 = (const float*)d_in[11];
  const float* lq2 = (const float*)d_in[12]; const float* lk2 = (const float*)d_in[13];
  const float* ln_g = (const float*)d_in[14]; const float* ln_b = (const float*)d_in[15];
  float* out = (float*)d_out;

  char* ws = (char*)d_ws;
  const size_t MB = 1ull << 20;
  _Float16* xh  = (_Float16*)(ws + 0);         //  8 MB: x as f16 [4096][1024]
  _Float16* WqT = (_Float16*)(ws + 8 * MB);    //  2 MB each, [N][K] f16
  _Float16* WkT = (_Float16*)(ws + 10 * MB);
  _Float16* WvT = (_Float16*)(ws + 12 * MB);
  _Float16* WoT = (_Float16*)(ws + 14 * MB);
  _Float16* qh  = (_Float16*)(ws + 16 * MB);   //  8 MB: [b][half][h][s][32]
  _Float16* kh  = (_Float16*)(ws + 24 * MB);   //  8 MB
  _Float16* vTh = (_Float16*)(ws + 32 * MB);   //  8 MB: [b][h][d][s]
  _Float16* aoh = (_Float16*)(ws + 40 * MB);   //  8 MB: [b][s][1024]
  // total workspace: 48 MB

  // 1) precision conversion + weight transposes
  cvt_f32_to_f16<<<(M_ROWS * HIDDEN_D + 255) / 256, 256, 0, stream>>>(
      x, xh, M_ROWS * HIDDEN_D);
  transpose_w<<<(HIDDEN_D * HIDDEN_D) / 256, 256, 0, stream>>>(Wq, WqT);
  transpose_w<<<(HIDDEN_D * HIDDEN_D) / 256, 256, 0, stream>>>(Wk, WkT);
  transpose_w<<<(HIDDEN_D * HIDDEN_D) / 256, 256, 0, stream>>>(Wv, WvT);
  transpose_w<<<(HIDDEN_D * HIDDEN_D) / 256, 256, 0, stream>>>(Wo, WoT);

  // 2) QKV projections: 3 * 64 * 32 macro-tiles, 4 waves/block
  qkv_gemm<<<(3 * 64 * 32) / 4, 128, 0, stream>>>(
      xh, WqT, WkT, WvT, bq, bk, bv, qh, kh, vTh);

  // 3) flash differential attention: 2*16*128 = 4096 tiles, 4 waves/block
  diff_attn<<<4096 / 4, 128, 0, stream>>>(
      qh, kh, vTh, mask, lq1, lk1, lq2, lk2, ln_g, ln_b, aoh);

  // 4) output projection: 64 * 32 macro-tiles
  out_proj<<<(64 * 32) / 4, 128, 0, stream>>>(aoh, WoT, bo, out);
}
